// FPInitializer_20469814133046
// MI455X (gfx1250) — matrix operand order repaired and checked by reading, not compiled, not run on hardware
//
#include <hip/hip_runtime.h>
#include <hip/hip_bf16.h>

// ---------------------------------------------------------------------------
// Problem shapes (fixed by the reference):
//   B=256, A=128, NB=256, D=16, FA=64, FB=32, FP=256
//   neighbor GEMM: M = B*A*D = 524288, K = 96, N = 256
//   atom     GEMM: M = B*A   =  32768, K = 64, N = 256
// Output: d_out = [atom_FP (32768x256) | neighbor_FP (524288x256)] f32
// Linear bias cancels exactly under BatchNorm (mean-subtraction) -> skipped.
// ---------------------------------------------------------------------------

typedef __attribute__((ext_vector_type(16))) __bf16 v16bf;
typedef __attribute__((ext_vector_type(8)))  float  v8f;

__device__ __forceinline__ unsigned short f32_to_bf16(float f) {
    unsigned int u = __float_as_uint(f);
    // round-to-nearest-even
    u += 0x7FFFu + ((u >> 16) & 1u);
    return (unsigned short)(u >> 16);
}

// ---------------------------------------------------------------------------
// Kernel 0: convert f32 weights -> bf16 workspace copies
// ---------------------------------------------------------------------------
__global__ __launch_bounds__(256) void convert_w_kernel(
    const float* __restrict__ neiW,  // [256,96]
    const float* __restrict__ atomW, // [256,64]
    unsigned short* __restrict__ neiWbf,
    unsigned short* __restrict__ atomWbf)
{
    int i = blockIdx.x * 256 + threadIdx.x;
    if (i < 256 * 96) neiWbf[i]  = f32_to_bf16(neiW[i]);
    if (i < 256 * 64) atomWbf[i] = f32_to_bf16(atomW[i]);
}

// ---------------------------------------------------------------------------
// WMMA GEMM: Y[m, c] = sum_k X[m,k] * W[c,k]   (bf16 inputs, f32 accumulate)
// Block = 256 threads (8 wave32), tile = 64 rows x 256 cols, K in steps of 32.
// GATHER=true : X row m is concat(atom_features[b, anl[m], :64],
//                                 bond_features[b, bnl[m], :32])
// GATHER=false: X = atomf, dense [M, KDIM]
// ---------------------------------------------------------------------------
template <int KDIM, bool GATHER>
__global__ __launch_bounds__(256) void gemm_wmma_kernel(
    const float* __restrict__ atomf,
    const float* __restrict__ bondf,
    const int*   __restrict__ anl,
    const int*   __restrict__ bnl,
    const unsigned short* __restrict__ Wbf,   // [256, KDIM] bf16 row-major
    float* __restrict__ Y)                    // [M, 256]
{
    constexpr int MT = 64;    // rows per block
    constexpr int XS = 112;   // LDS row stride in bf16 elems (mult of 8 -> 16B aligned rows)
    __shared__ alignas(16) unsigned short sX[MT * XS];

    const int t      = threadIdx.x;
    const int m_base = blockIdx.x * MT;

    // ---- stage X tile (f32 -> bf16) into LDS ----
    {
        const int r = t >> 2;       // row 0..63
        const int q = t & 3;        // quarter of the K range
        unsigned short* dst = sX + r * XS;
        if (GATHER) {
            const int m  = m_base + r;
            const int b  = m >> 11;          // / (A*D) = /2048
            const int a  = (m >> 4) & 127;   // (m/D) % A
            const int d  = m & 15;
            const int ia = anl[(b * 128 + a) * 16 + d];
            const int ib = bnl[(b * 128 + a) * 16 + d];
            const float* arow = atomf + ((long)b * 128 + ia) * 64;
            const float* brow = bondf + ((long)b * 256 + ib) * 32;
#pragma unroll
            for (int j = 0; j < 24; ++j) {           // 4 * 24 = 96 = KDIM
                int k = q * 24 + j;
                float x = (k < 64) ? arow[k] : brow[k - 64];
                dst[k] = f32_to_bf16(x);
            }
        } else {
            const float* xrow = atomf + (long)(m_base + r) * KDIM;
#pragma unroll
            for (int j = 0; j < 16; ++j) {           // 4 * 16 = 64 = KDIM
                int k = q * 16 + j;
                dst[k] = f32_to_bf16(xrow[k]);
            }
        }
    }
    __syncthreads();

    const int wave  = t >> 5;
    const int lane  = t & 31;
    const int lhalf = lane >> 4;   // 0: lanes 0-15, 1: lanes 16-31
    const int l16   = lane & 15;
    const int mi    = wave & 3;           // M subtile (16 rows) 0..3
    const int nj0   = (wave >> 2) * 8;    // first of 8 owned N subtiles

    v8f acc[8];
#pragma unroll
    for (int j = 0; j < 8; ++j)
        acc[j] = v8f{0.f, 0.f, 0.f, 0.f, 0.f, 0.f, 0.f, 0.f};

    constexpr int KSTEPS = KDIM / 32;
#pragma unroll
    for (int ks = 0; ks < KSTEPS; ++ks) {
        // A fragment (16x32 bf16) from LDS: per ISA layout, slots v0..3 are
        // 8 consecutive K at (ks*32 + 8*lhalf), slots v4..7 at +16.
        union { v16bf v; uint4 u[2]; } A;
        const unsigned short* ap = sX + (mi * 16 + l16) * XS + ks * 32 + lhalf * 8;
        A.u[0] = *(const uint4*)(ap);
        A.u[1] = *(const uint4*)(ap + 16);

#pragma unroll
        for (int j = 0; j < 8; ++j) {
            const int n = (nj0 + j) * 16 + l16;  // output channel
            // B fragment (32x16 bf16): lane holds col n, 8 consecutive K at
            // (ks*32 + 8*lhalf) and 8 more at +16, matching A.
            union { v16bf v; uint4 u[2]; } Bf;
            const unsigned short* bp = Wbf + n * KDIM + ks * 32 + lhalf * 8;
            Bf.u[0] = *(const uint4*)(bp);
            Bf.u[1] = *(const uint4*)(bp + 16);

            acc[j] = __builtin_amdgcn_wmma_f32_16x16x32_bf16(
                false, A.v, false, Bf.v, (short)0, acc[j], false, false);
        }
    }

    // ---- store D: element [M = v + 8*lhalf, N = l16] of each 16x16 tile ----
#pragma unroll
    for (int j = 0; j < 8; ++j) {
        const int n = (nj0 + j) * 16 + l16;
#pragma unroll
        for (int v = 0; v < 8; ++v) {
            long row = (long)(m_base + mi * 16 + lhalf * 8 + v);
            Y[row * 256 + n] = acc[j][v];
        }
    }
}

// ---------------------------------------------------------------------------
// Per-chunk column partial sums (deterministic: one thread per channel,
// sequential accumulation over a fixed row range).
// ---------------------------------------------------------------------------
__global__ __launch_bounds__(256) void col_stats_partial_kernel(
    const float* __restrict__ Y, int rows_per_chunk,
    float* __restrict__ psum, float* __restrict__ psq)
{
    const int c = threadIdx.x;
    const long r0 = (long)blockIdx.x * rows_per_chunk;
    const float* p = Y + r0 * 256 + c;
    float s = 0.f, q = 0.f;
    for (int i = 0; i < rows_per_chunk; ++i) {
        float y = p[(long)i * 256];
        s += y;
        q += y * y;
    }
    psum[blockIdx.x * 256 + c] = s;
    psq [blockIdx.x * 256 + c] = q;
}

__global__ __launch_bounds__(256) void col_stats_final_kernel(
    const float* __restrict__ psum, const float* __restrict__ psq,
    int nchunks, float inv_rows,
    const float* __restrict__ gamma, const float* __restrict__ beta,
    float* __restrict__ scale, float* __restrict__ shift)
{
    const int c = threadIdx.x;
    float s = 0.f, q = 0.f;
    for (int i = 0; i < nchunks; ++i) {
        s += psum[i * 256 + c];
        q += psq [i * 256 + c];
    }
    float mu  = s * inv_rows;
    float var = q * inv_rows - mu * mu;            // biased variance (matches jnp.var)
    float sc  = gamma[c] * rsqrtf(var + 1e-6f);
    scale[c] = sc;
    shift[c] = beta[c] - mu * sc;
}

// ---------------------------------------------------------------------------
// In-place BN affine + LeakyReLU(0.01), float4-vectorized (4 channels/elem).
// ---------------------------------------------------------------------------
__global__ __launch_bounds__(256) void bn_act_kernel(
    float* __restrict__ Y, long n4,
    const float* __restrict__ scale, const float* __restrict__ shift)
{
    __shared__ float ssc[256], ssh[256];
    ssc[threadIdx.x] = scale[threadIdx.x];
    ssh[threadIdx.x] = shift[threadIdx.x];
    __syncthreads();

    long i = (long)blockIdx.x * 256 + threadIdx.x;
    const long stride = (long)gridDim.x * 256;
    float4* Y4 = (float4*)Y;
    for (; i < n4; i += stride) {
        float4 y = Y4[i];
        int c4 = (int)(i & 63) * 4;  // (i*4) % 256
        float a = y.x * ssc[c4 + 0] + ssh[c4 + 0];
        float b = y.y * ssc[c4 + 1] + ssh[c4 + 1];
        float c = y.z * ssc[c4 + 2] + ssh[c4 + 2];
        float d = y.w * ssc[c4 + 3] + ssh[c4 + 3];
        y.x = (a >= 0.f) ? a : 0.01f * a;
        y.y = (b >= 0.f) ? b : 0.01f * b;
        y.z = (c >= 0.f) ? c : 0.01f * c;
        y.w = (d >= 0.f) ? d : 0.01f * d;
        Y4[i] = y;
    }
}

// ---------------------------------------------------------------------------
extern "C" void kernel_launch(void* const* d_in, const int* in_sizes, int n_in,
                              void* d_out, int out_size, void* d_ws, size_t ws_size,
                              hipStream_t stream)
{
    const float* atomf      = (const float*)d_in[0];   // [256,128,64]
    const float* bondf      = (const float*)d_in[1];   // [256,256,32]
    const int*   anl        = (const int*)  d_in[2];   // [256,128,16]
    const int*   bnl        = (const int*)  d_in[3];   // [256,128,16]
    const float* atom_W     = (const float*)d_in[4];   // [256,64]
    // d_in[5] atom_b: exactly cancelled by BN mean subtraction
    const float* atom_gamma = (const float*)d_in[6];
    const float* atom_beta  = (const float*)d_in[7];
    const float* nei_W      = (const float*)d_in[8];   // [256,96]
    // d_in[9] nei_b: cancelled
    const float* nei_gamma  = (const float*)d_in[10];
    const float* nei_beta   = (const float*)d_in[11];

    const long M_ATOM = 32768;   // 256*128
    const long M_NEI  = 524288;  // 256*128*16

    float* outAtom = (float*)d_out;                     // [32768 , 256]
    float* outNei  = (float*)d_out + M_ATOM * 256;      // [524288, 256]

    // ---- workspace layout ----
    char* ws = (char*)d_ws;
    unsigned short* WbfNei  = (unsigned short*)(ws);            // 256*96*2  = 49152 B
    unsigned short* WbfAtom = (unsigned short*)(ws + 49152);    // 256*64*2  = 32768 B
    float* psumN = (float*)(ws + 81920);                        // 512*256*4
    float* psqN  = (float*)(ws + 81920 + 524288);               // 512*256*4
    float* psumA = (float*)(ws + 81920 + 1048576);              //  64*256*4
    float* psqA  = (float*)(ws + 81920 + 1048576 + 65536);      //  64*256*4
    float* scN   = (float*)(ws + 81920 + 1048576 + 131072);
    float* shN   = scN + 256;
    float* scA   = shN + 256;
    float* shA   = scA + 256;

    // 1) weights -> bf16
    convert_w_kernel<<<96, 256, 0, stream>>>(nei_W, atom_W, WbfNei, WbfAtom);

    // 2) WMMA GEMMs (pre-BN y, bias skipped)
    gemm_wmma_kernel<96, true ><<<8192, 256, 0, stream>>>(atomf, bondf, anl, bnl, WbfNei, outNei);
    gemm_wmma_kernel<64, false><<< 512, 256, 0, stream>>>(atomf, nullptr, nullptr, nullptr, WbfAtom, outAtom);

    // 3) batch statistics (deterministic two-stage reduction)
    col_stats_partial_kernel<<<512, 256, 0, stream>>>(outNei, 1024, psumN, psqN);
    col_stats_partial_kernel<<< 64, 256, 0, stream>>>(outAtom, 512, psumA, psqA);
    col_stats_final_kernel<<<1, 256, 0, stream>>>(psumN, psqN, 512, 1.0f / (float)M_NEI,
                                                  nei_gamma, nei_beta, scN, shN);
    col_stats_final_kernel<<<1, 256, 0, stream>>>(psumA, psqA, 64, 1.0f / (float)M_ATOM,
                                                  atom_gamma, atom_beta, scA, shA);

    // 4) BN affine + LeakyReLU, in place
    bn_act_kernel<<<8192, 256, 0, stream>>>(outNei, (M_NEI * 256) / 4, scN, shN);
    bn_act_kernel<<<2048, 256, 0, stream>>>(outAtom, (M_ATOM * 256) / 4, scA, shA);
}
